// STEncoderBlock_13520557048169
// MI455X (gfx1250) — matrix operand order repaired and checked by reading, not compiled, not run on hardware
//
#include <hip/hip_runtime.h>
#include <math.h>

// ---------------------------------------------------------------------------
// Problem constants (from reference): B=16, N=1024, D=128, HEADS=4, HEAD_DIM=16
// GEO_DIM=64, EMB=16, CHEB_K=2, SCALE=0.25
// ---------------------------------------------------------------------------
#define LN_EPS 1e-5f

typedef __attribute__((ext_vector_type(16))) __bf16 v16bf;
typedef __attribute__((ext_vector_type(8)))  float  v8f;

static __device__ __forceinline__ int lane_n() { return threadIdx.x & 15; }
static __device__ __forceinline__ int lane_h() { return (threadIdx.x >> 4) & 1; }

static __device__ __forceinline__ __bf16 f2bf(float f) {
  unsigned u = __builtin_bit_cast(unsigned, f);
  u += 0x7fffu + ((u >> 16) & 1u);                 // round-to-nearest-even
  unsigned short h = (unsigned short)(u >> 16);
  return __builtin_bit_cast(__bf16, h);
}

static __device__ __forceinline__ v8f zero8() {
  v8f z;
#pragma unroll
  for (int i = 0; i < 8; ++i) z[i] = 0.f;
  return z;
}

static __device__ __forceinline__ v8f wmma_bf16(v16bf a, v16bf b, v8f c) {
  // D = A(16x32) * B(32x16) + C, fp32 accumulate
  return __builtin_amdgcn_wmma_f32_16x16x32_bf16(false, a, false, b, (short)0, c,
                                                 false, false);
}

// ---- fragment loaders -----------------------------------------------------
// A-fragment: lane m = lane_n(); element e -> K = k0 + half*16 + e
static __device__ __forceinline__ v16bf ldfragA(const __bf16* rowp, int k0) {
  v16bf f;
  const __bf16* p = rowp + k0 + lane_h() * 16;
  *(uint4*)&f       = *(const uint4*)(p);
  *((uint4*)&f + 1) = *(const uint4*)(p + 8);
  return f;
}
// A-fragment with only K=0..15 valid (zero pad K=16..31)
static __device__ __forceinline__ v16bf ldfragA16(const __bf16* rowp) {
  v16bf f;
  uint4 z = make_uint4(0u, 0u, 0u, 0u);
  if (lane_h() == 0) {
    *(uint4*)&f       = *(const uint4*)(rowp);
    *((uint4*)&f + 1) = *(const uint4*)(rowp + 8);
  } else {
    *(uint4*)&f       = z;
    *((uint4*)&f + 1) = z;
  }
  return f;
}
static __device__ __forceinline__ v16bf ldfragA16_f32(const float* rowp) {
  v16bf f;
  bool lo = (lane_h() == 0);
#pragma unroll
  for (int e = 0; e < 16; ++e) f[e] = f2bf(lo ? rowp[e] : 0.f);
  return f;
}
// B-fragment: lane col = lane_n(); element e -> K = k0 + half*16 + e
static __device__ __forceinline__ v16bf ldfragB(const __bf16* rowp, int k0) {
  v16bf f;
  const uint4* q = (const uint4*)(rowp + k0 + lane_h() * 16);
  *(uint4*)&f       = q[0];
  *((uint4*)&f + 1) = q[1];
  return f;
}
// B-fragment with only K=0..15 valid
static __device__ __forceinline__ v16bf ldfragB16(const __bf16* rowp) {
  v16bf f;
  uint4 z = make_uint4(0u, 0u, 0u, 0u);
  if (lane_h() == 0) {
    const uint4* q = (const uint4*)rowp;
    *(uint4*)&f       = q[0];
    *((uint4*)&f + 1) = q[1];
  } else {
    *(uint4*)&f       = z;
    *((uint4*)&f + 1) = z;
  }
  return f;
}
static __device__ __forceinline__ v16bf ldfragB16_f32(const float* rowp) {
  v16bf f;
  if (lane_h() == 0) {
#pragma unroll
    for (int e = 0; e < 16; ++e) f[e] = f2bf(rowp[e]);
  } else {
#pragma unroll
    for (int e = 0; e < 16; ++e) f[e] = f2bf(0.f);
  }
  return f;
}

// ---------------------------------------------------------------------------
// elementwise f32 -> bf16 convert
__global__ void k_cvt(const float* __restrict__ x, __bf16* __restrict__ y, int n) {
  int i = blockIdx.x * blockDim.x + threadIdx.x;
  if (i < n) y[i] = f2bf(x[i]);
}

// WpT[c][d] = Wp[d][c], c in [0,32768), d in [0,16)  (bf16 out)
__global__ void k_wpt(const float* __restrict__ Wp, __bf16* __restrict__ WpT) {
  int c = blockIdx.x * blockDim.x + threadIdx.x;
  __bf16 t[16];
#pragma unroll
  for (int d = 0; d < 16; ++d) t[d] = f2bf(Wp[d * 32768 + c]);
  __bf16* dst = WpT + (size_t)c * 16;
  *(uint4*)dst       = *(uint4*)t;
  *((uint4*)dst + 1) = *((uint4*)t + 1);
}

// zT[b][o][n] = z[b][n][o]  (bf16); grid = B*64 blocks x 128 threads
__global__ void k_ztr(const float* __restrict__ z, __bf16* __restrict__ zT) {
  int b = blockIdx.x >> 6, nt = blockIdx.x & 63, o = threadIdx.x;
  __bf16 t[16];
#pragma unroll
  for (int i = 0; i < 16; ++i)
    t[i] = f2bf(z[((size_t)b * 1024 + nt * 16 + i) * 128 + o]);
  __bf16* dst = zT + ((size_t)(b * 128 + o)) * 1024 + nt * 16;
  *(uint4*)dst       = *(uint4*)t;
  *((uint4*)dst + 1) = *((uint4*)t + 1);
}

// LayerNorm over 128 features, one wave per row, bf16 output
__global__ void k_ln(const float* __restrict__ x, const float* __restrict__ g,
                     const float* __restrict__ bb, __bf16* __restrict__ y, int rows) {
  int wave = threadIdx.x >> 5, lane = threadIdx.x & 31;
  int row = blockIdx.x * (blockDim.x >> 5) + wave;
  if (row >= rows) return;
  const float* xr = x + (size_t)row * 128;
  float v0 = xr[lane], v1 = xr[lane + 32], v2 = xr[lane + 64], v3 = xr[lane + 96];
  float s = v0 + v1 + v2 + v3;
  float q = v0 * v0 + v1 * v1 + v2 * v2 + v3 * v3;
#pragma unroll
  for (int o = 16; o >= 1; o >>= 1) {
    s += __shfl_xor(s, o, 32);
    q += __shfl_xor(q, o, 32);
  }
  float mean = s * (1.f / 128.f);
  float var  = q * (1.f / 128.f) - mean * mean;
  float rstd = rsqrtf(var + LN_EPS);
  __bf16* yr = y + (size_t)row * 128;
  yr[lane]      = f2bf((v0 - mean) * rstd * g[lane]      + bb[lane]);
  yr[lane + 32] = f2bf((v1 - mean) * rstd * g[lane + 32] + bb[lane + 32]);
  yr[lane + 64] = f2bf((v2 - mean) * rstd * g[lane + 64] + bb[lane + 64]);
  yr[lane + 96] = f2bf((v3 - mean) * rstd * g[lane + 96] + bb[lane + 96]);
}

// ---------------------------------------------------------------------------
// QKV projections: 6 projs x 1024 token-blocks, one wave each.
// q,k stored [attn][b][h][n][16]; v stored transposed [attn][b][h][16][n]
__global__ void k_qkv(const __bf16* __restrict__ hn, const __bf16* __restrict__ Wall,
                      const float* b0, const float* b1, const float* b2,
                      const float* b3, const float* b4, const float* b5,
                      __bf16* __restrict__ qbuf, __bf16* __restrict__ kbuf,
                      __bf16* __restrict__ vT) {
  int w = (blockIdx.x * blockDim.x + threadIdx.x) >> 5;  // 0..6143
  int p = w >> 10, blk = w & 1023;
  const __bf16* W = Wall + (size_t)p * 64 * 128;
  const float* bias = (p == 0) ? b0 : (p == 1) ? b1 : (p == 2) ? b2
                    : (p == 3) ? b3 : (p == 4) ? b4 : b5;
  const __bf16* arow = hn + ((size_t)(blk * 16 + lane_n())) * 128;
  v8f acc[4];
#pragma unroll
  for (int ot = 0; ot < 4; ++ot) acc[ot] = zero8();
#pragma unroll
  for (int k0 = 0; k0 < 128; k0 += 32) {
    v16bf af = ldfragA(arow, k0);
#pragma unroll
    for (int ot = 0; ot < 4; ++ot)
      acc[ot] = wmma_bf16(af, ldfragB(W + (size_t)(ot * 16 + lane_n()) * 128, k0), acc[ot]);
  }
  int attn = p / 3, kind = p % 3;
#pragma unroll
  for (int ot = 0; ot < 4; ++ot) {
    int o = ot * 16 + lane_n(), h = o >> 4, od = o & 15;
    float bv = bias[o];
    if (kind < 2) {
      __bf16* dst = (kind == 0) ? qbuf : kbuf;
#pragma unroll
      for (int r = 0; r < 8; ++r) {
        int tok = blk * 16 + r + 8 * lane_h();
        int b = tok >> 10, n = tok & 1023;
        dst[(((size_t)(attn * 16 + b) * 4 + h) * 1024 + n) * 16 + od] =
            f2bf(acc[ot][r] + bv);
      }
    } else {
      __bf16 pk[8];
#pragma unroll
      for (int r = 0; r < 8; ++r) pk[r] = f2bf(acc[ot][r] + bv);
      int tok0 = blk * 16 + 8 * lane_h();
      int b = tok0 >> 10, n0 = tok0 & 1023;
      __bf16* dst = vT + (((size_t)(attn * 16 + b) * 4 + h) * 16 + od) * 1024 + n0;
      *(uint4*)dst = *(uint4*)pk;
    }
  }
}

// ---------------------------------------------------------------------------
// Flash attention. One wave per (attn, b, h, 16-token block). Computes S^T =
// K*Q^T so softmax rows are lane-local; P^T tile feeds the O^T WMMA as a
// B-fragment with a single cross-half shuffle.
__global__ void k_attn(const __bf16* __restrict__ qbuf, const __bf16* __restrict__ kbuf,
                       const __bf16* __restrict__ vT,
                       const unsigned char* __restrict__ geo_mask,
                       const unsigned char* __restrict__ sem_mask,
                       __bf16* __restrict__ attn_out) {
  int w = (blockIdx.x * blockDim.x + threadIdx.x) >> 5;  // 0..8191
  int nb = w & 63, h = (w >> 6) & 3, b = (w >> 8) & 15, a = w >> 12;
  const unsigned char* mask = a ? sem_mask : geo_mask;
  size_t bh = (size_t)(a * 16 + b) * 4 + h;
  const __bf16* qb = qbuf + bh * 1024 * 16;
  const __bf16* kb = kbuf + bh * 1024 * 16;
  const __bf16* vb = vT + bh * 16 * 1024;

  v16bf qf = ldfragB16(qb + (size_t)(nb * 16 + lane_n()) * 16);
  const unsigned char* mrow = mask + (size_t)(nb * 16 + lane_n()) * 1024;

  float mrun = -__builtin_inff(), l = 0.f;
  v8f oacc = zero8();
  int hh = lane_h();

  for (int mb = 0; mb < 32; ++mb) {
    int m0 = mb * 32;
    v16bf a0 = ldfragA16(kb + (size_t)(m0 + lane_n()) * 16);
    v16bf a1 = ldfragA16(kb + (size_t)(m0 + 16 + lane_n()) * 16);
    v8f c0 = wmma_bf16(a0, qf, zero8());
    v8f c1 = wmma_bf16(a1, qf, zero8());
    unsigned long long mk0 = *(const unsigned long long*)(mrow + m0 + 8 * hh);
    unsigned long long mk1 = *(const unsigned long long*)(mrow + m0 + 16 + 8 * hh);
    float s0[8], s1[8], tmax = -1e30f;
#pragma unroll
    for (int r = 0; r < 8; ++r) {
      s0[r] = ((mk0 >> (8 * r)) & 0xff) ? -1e30f : c0[r] * 0.25f;
      s1[r] = ((mk1 >> (8 * r)) & 0xff) ? -1e30f : c1[r] * 0.25f;
      tmax = fmaxf(tmax, fmaxf(s0[r], s1[r]));
    }
    tmax = fmaxf(tmax, __shfl_xor(tmax, 16, 32));
    float mnew  = fmaxf(mrun, tmax);
    float alpha = __expf(mrun - mnew);
    float p0[8], p1[8], ts = 0.f;
#pragma unroll
    for (int r = 0; r < 8; ++r) {
      p0[r] = __expf(s0[r] - mnew);
      p1[r] = __expf(s1[r] - mnew);
      ts += p0[r] + p1[r];
    }
    ts += __shfl_xor(ts, 16, 32);
    l = l * alpha + ts;
    mrun = mnew;
#pragma unroll
    for (int i = 0; i < 8; ++i) oacc[i] *= alpha;
    float q0[8], q1[8];
#pragma unroll
    for (int r = 0; r < 8; ++r) {
      q0[r] = __shfl_xor(p0[r], 16, 32);
      q1[r] = __shfl_xor(p1[r], 16, 32);
    }
    v16bf pb;
#pragma unroll
    for (int e = 0; e < 8; ++e)  pb[e] = f2bf(hh ? q1[e] : p0[e]);
#pragma unroll
    for (int e = 8; e < 16; ++e) pb[e] = f2bf(hh ? p1[e - 8] : q0[e - 8]);
    v16bf vf = ldfragA(vb + (size_t)lane_n() * 1024, m0);
    oacc = wmma_bf16(vf, pb, oacc);  // O^T (16 o x 16 n)
  }
  float inv = 1.f / l;
  __bf16 pk[8];
#pragma unroll
  for (int r = 0; r < 8; ++r) pk[r] = f2bf(oacc[r] * inv);
  int tok = nb * 16 + lane_n();
  __bf16* dst = attn_out + ((size_t)(b * 1024 + tok)) * 128 + a * 64 + h * 16 + 8 * hh;
  *(uint4*)dst = *(uint4*)pk;
}

// ---------------------------------------------------------------------------
// A_sup = row-softmax(relu(E E^T)); G is symmetric so the computed tile's
// column index == softmax row -> lane-local reductions. 64 waves.
__global__ void k_asup(const float* __restrict__ E, __bf16* __restrict__ Asup) {
  int nb = (blockIdx.x * blockDim.x + threadIdx.x) >> 5;  // 0..63
  v16bf bf = ldfragB16_f32(E + (size_t)(nb * 16 + lane_n()) * 16);
  float rmax = 0.f;  // relu'd values are >= 0
  for (int mt = 0; mt < 64; ++mt) {
    v16bf af = ldfragA16_f32(E + (size_t)(mt * 16 + lane_n()) * 16);
    v8f c = wmma_bf16(af, bf, zero8());
#pragma unroll
    for (int r = 0; r < 8; ++r) rmax = fmaxf(rmax, fmaxf(c[r], 0.f));
  }
  rmax = fmaxf(rmax, __shfl_xor(rmax, 16, 32));
  float ls = 0.f;
  for (int mt = 0; mt < 64; ++mt) {
    v16bf af = ldfragA16_f32(E + (size_t)(mt * 16 + lane_n()) * 16);
    v8f c = wmma_bf16(af, bf, zero8());
#pragma unroll
    for (int r = 0; r < 8; ++r) ls += __expf(fmaxf(c[r], 0.f) - rmax);
  }
  ls += __shfl_xor(ls, 16, 32);
  float inv = 1.f / ls;
  for (int mt = 0; mt < 64; ++mt) {
    v16bf af = ldfragA16_f32(E + (size_t)(mt * 16 + lane_n()) * 16);
    v8f c = wmma_bf16(af, bf, zero8());
    __bf16 pk[8];
#pragma unroll
    for (int r = 0; r < 8; ++r) pk[r] = f2bf(__expf(fmaxf(c[r], 0.f) - rmax) * inv);
    __bf16* dst = Asup + (size_t)(nb * 16 + lane_n()) * 1024 + mt * 16 + 8 * lane_h();
    *(uint4*)dst = *(uint4*)pk;
  }
}

// zA[b] = A_sup @ z[b]; one wave per (b, 16-row block). 256 WMMA/wave.
__global__ void k_za(const __bf16* __restrict__ Asup, const __bf16* __restrict__ zT,
                     __bf16* __restrict__ zA) {
  int w = (blockIdx.x * blockDim.x + threadIdx.x) >> 5;  // 0..1023
  int b = w >> 6, nb = w & 63;
  const __bf16* arow = Asup + (size_t)(nb * 16 + lane_n()) * 1024;
  v8f acc[8];
#pragma unroll
  for (int ot = 0; ot < 8; ++ot) acc[ot] = zero8();
  for (int k0 = 0; k0 < 1024; k0 += 32) {
    v16bf af = ldfragA(arow, k0);
#pragma unroll
    for (int ot = 0; ot < 8; ++ot) {
      const __bf16* brow = zT + ((size_t)(b * 128 + ot * 16 + lane_n())) * 1024;
      acc[ot] = wmma_bf16(af, ldfragB(brow, k0), acc[ot]);
    }
  }
#pragma unroll
  for (int ot = 0; ot < 8; ++ot) {
    int o = ot * 16 + lane_n();
#pragma unroll
    for (int r = 0; r < 8; ++r) {
      int n = nb * 16 + r + 8 * lane_h();
      zA[((size_t)(b * 1024 + n)) * 128 + o] = f2bf(acc[ot][r]);
    }
  }
}

// ---------------------------------------------------------------------------
// ChebNet apply: per 16-node group, stage1 builds per-node weights
// W_node = E[n] . Wp via WMMA into LDS (bf16, padded stride 40), stage2 does
// z_g(16 batches x 256) @ W_node(256x128) per node. 8 waves cooperate.
__global__ void k_cheb(const float* __restrict__ E, const __bf16* __restrict__ WpT,
                       const __bf16* __restrict__ zbf, const __bf16* __restrict__ zabf,
                       const float* __restrict__ bias_pool, float* __restrict__ zstat) {
  extern __shared__ __bf16 Lw[];  // [16 nodes][128 o][40] (ki block of 32 + pad)
  int grp = blockIdx.x, n0 = grp * 16;
  int wv = threadIdx.x >> 5;
  v16bf eA = ldfragA16_f32(E + (size_t)(n0 + lane_n()) * 16);  // nodes x d
  v8f acc[2][8];
#pragma unroll
  for (int s = 0; s < 2; ++s)
#pragma unroll
    for (int ot = 0; ot < 8; ++ot) acc[s][ot] = zero8();

  for (int kb = 0; kb < 8; ++kb) {
    __syncthreads();
    // stage1: 256 col-tiles (32 ki x 8 o-tiles); 32 per wave
    for (int i = 0; i < 32; ++i) {
      int t = wv * 32 + i;
      int kil = t >> 3, ot = t & 7;
      int kig = kb * 32 + kil;
      int o = ot * 16 + lane_n();
      v16bf bfr = ldfragB16(WpT + ((size_t)(kig * 128 + o)) * 16);
      v8f c = wmma_bf16(eA, bfr, zero8());
#pragma unroll
      for (int r = 0; r < 8; ++r) {
        int node = r + 8 * lane_h();
        Lw[((size_t)(node * 128 + o)) * 40 + kil] = f2bf(c[r]);
      }
    }
    __syncthreads();
    // stage2: each wave owns nodes {2wv, 2wv+1}
    bool useZ = (kb < 4);
    int kloc = kb * 32 - (useZ ? 0 : 128);
    const __bf16* src = useZ ? zbf : zabf;
#pragma unroll
    for (int s = 0; s < 2; ++s) {
      int nl = wv * 2 + s;
      int n = n0 + nl;
      const __bf16* arow = src + ((size_t)(lane_n() * 1024 + n)) * 128 + kloc;
      v16bf af = ldfragA(arow, 0);  // lane = batch
#pragma unroll
      for (int ot = 0; ot < 8; ++ot) {
        const __bf16* lp = &Lw[((size_t)(nl * 128 + ot * 16 + lane_n())) * 40];
        acc[s][ot] = wmma_bf16(af, ldfragB(lp, 0), acc[s][ot]);
      }
    }
  }
#pragma unroll
  for (int s = 0; s < 2; ++s) {
    int nl = wv * 2 + s, n = n0 + nl;
#pragma unroll
    for (int ot = 0; ot < 8; ++ot) {
      int o = ot * 16 + lane_n();
      float bv = 0.f;
#pragma unroll
      for (int d = 0; d < 16; ++d) bv += E[(size_t)n * 16 + d] * bias_pool[d * 128 + o];
#pragma unroll
      for (int r = 0; r < 8; ++r) {
        int b = r + 8 * lane_h();
        zstat[((size_t)(b * 1024) + n) * 128 + o] = acc[s][ot][r] + bv;
      }
    }
  }
}

// ---------------------------------------------------------------------------
// Generic  Y = [gelu]( X(Mx128) @ W^T(128x128) + bias ) [+ addsrc]; one wave
// per 16-row block. Optional f32 and/or bf16 outputs.
__global__ void k_gemm128(const __bf16* __restrict__ X, const __bf16* __restrict__ W,
                          const float* __restrict__ bias, const float* __restrict__ addsrc,
                          float* __restrict__ outf, __bf16* __restrict__ outb,
                          int do_gelu) {
  int tb = (blockIdx.x * blockDim.x + threadIdx.x) >> 5;  // 0..1023
  const __bf16* arow = X + ((size_t)(tb * 16 + lane_n())) * 128;
  v8f acc[8];
#pragma unroll
  for (int ot = 0; ot < 8; ++ot) acc[ot] = zero8();
#pragma unroll
  for (int k0 = 0; k0 < 128; k0 += 32) {
    v16bf af = ldfragA(arow, k0);
#pragma unroll
    for (int ot = 0; ot < 8; ++ot)
      acc[ot] = wmma_bf16(af, ldfragB(W + (size_t)(ot * 16 + lane_n()) * 128, k0), acc[ot]);
  }
#pragma unroll
  for (int ot = 0; ot < 8; ++ot) {
    int o = ot * 16 + lane_n();
    float bv = bias[o];
#pragma unroll
    for (int r = 0; r < 8; ++r) {
      int tok = tb * 16 + r + 8 * lane_h();
      float v = acc[ot][r] + bv;
      if (do_gelu) v = 0.5f * v * (1.f + erff(v * 0.70710678118654752f));
      if (addsrc) v += addsrc[(size_t)tok * 128 + o];
      if (outf) outf[(size_t)tok * 128 + o] = v;
      if (outb) outb[(size_t)tok * 128 + o] = f2bf(v);
    }
  }
}

// ---------------------------------------------------------------------------
extern "C" void kernel_launch(void* const* d_in, const int* in_sizes, int n_in,
                              void* d_out, int out_size, void* d_ws, size_t ws_size,
                              hipStream_t stream) {
  (void)in_sizes; (void)n_in; (void)out_size; (void)ws_size;
  const float* z   = (const float*)d_in[0];
  const float* h   = (const float*)d_in[1];
  const unsigned char* geo_mask = (const unsigned char*)d_in[2];
  const unsigned char* sem_mask = (const unsigned char*)d_in[3];
  // params in dict insertion order
  const float* geo_q_W = (const float*)d_in[4];  const float* geo_q_b = (const float*)d_in[5];
  const float* geo_k_W = (const float*)d_in[6];  const float* geo_k_b = (const float*)d_in[7];
  const float* geo_v_W = (const float*)d_in[8];  const float* geo_v_b = (const float*)d_in[9];
  const float* sem_q_W = (const float*)d_in[10]; const float* sem_q_b = (const float*)d_in[11];
  const float* sem_k_W = (const float*)d_in[12]; const float* sem_k_b = (const float*)d_in[13];
  const float* sem_v_W = (const float*)d_in[14]; const float* sem_v_b = (const float*)d_in[15];
  const float* proj_W  = (const float*)d_in[16]; const float* proj_b  = (const float*)d_in[17];
  const float* amlp_W1 = (const float*)d_in[18]; const float* amlp_b1 = (const float*)d_in[19];
  const float* amlp_W2 = (const float*)d_in[20]; const float* amlp_b2 = (const float*)d_in[21];
  const float* bmlp_W1 = (const float*)d_in[22]; const float* bmlp_b1 = (const float*)d_in[23];
  const float* bmlp_W2 = (const float*)d_in[24]; const float* bmlp_b2 = (const float*)d_in[25];
  const float* node_emb    = (const float*)d_in[26];
  const float* weights_pool= (const float*)d_in[27];
  const float* bias_pool   = (const float*)d_in[28];
  const float* ln1_g = (const float*)d_in[29]; const float* ln1_b = (const float*)d_in[30];
  const float* ln2_g = (const float*)d_in[31]; const float* ln2_b = (const float*)d_in[32];

  char* ws = (char*)d_ws;
  const size_t MB = 1ull << 20;
  __bf16* hn    = (__bf16*)(ws + 0 * MB);    // later reused as hn2
  __bf16* qbuf  = (__bf16*)(ws + 4 * MB);    // later reused as proj_out
  __bf16* kbuf  = (__bf16*)(ws + 8 * MB);    // later reused as amlp_h
  __bf16* vT    = (__bf16*)(ws + 12 * MB);   // later reused as bmlp_h
  __bf16* attn  = (__bf16*)(ws + 16 * MB);
  __bf16* zbf   = (__bf16*)(ws + 20 * MB);
  __bf16* zT    = (__bf16*)(ws + 24 * MB);
  __bf16* zA    = (__bf16*)(ws + 28 * MB);
  __bf16* Asup  = (__bf16*)(ws + 32 * MB);
  __bf16* Wbf   = (__bf16*)(ws + 34 * MB);   // 131072 bf16 weights
  __bf16* WpT   = (__bf16*)(ws + 35 * MB);   // 32768 x 16
  float*  zstat = (float*)(ws + 36 * MB);    // 8 MB
  float*  z1    = (float*)(ws + 44 * MB);    // 8 MB
  __bf16* proj_out = qbuf;
  __bf16* amlp_h   = kbuf;
  __bf16* hn2      = hn;
  __bf16* bmlp_h   = vT;
  float*  out      = (float*)d_out;

  // ---- weight / activation conversion ----
  const float* wsrc[11] = {geo_q_W, geo_k_W, geo_v_W, sem_q_W, sem_k_W, sem_v_W,
                           proj_W, amlp_W1, amlp_W2, bmlp_W1, bmlp_W2};
  const int    woff[11] = {0, 8192, 16384, 24576, 32768, 40960,
                           49152, 65536, 81920, 98304, 114688};
  const int    wcnt[11] = {8192, 8192, 8192, 8192, 8192, 8192,
                           16384, 16384, 16384, 16384, 16384};
  for (int i = 0; i < 11; ++i)
    k_cvt<<<(wcnt[i] + 255) / 256, 256, 0, stream>>>(wsrc[i], Wbf + woff[i], wcnt[i]);
  k_cvt<<<(2097152 + 255) / 256, 256, 0, stream>>>(z, zbf, 2097152);
  k_wpt<<<128, 256, 0, stream>>>(weights_pool, WpT);
  k_ztr<<<1024, 128, 0, stream>>>(z, zT);

  // ---- hn = LN1(h) ----
  k_ln<<<2048, 256, 0, stream>>>(h, ln1_g, ln1_b, hn, 16384);

  // ---- attention path ----
  k_qkv<<<768, 256, 0, stream>>>(hn, Wbf, geo_q_b, geo_k_b, geo_v_b,
                                 sem_q_b, sem_k_b, sem_v_b, qbuf, kbuf, vT);
  k_attn<<<1024, 256, 0, stream>>>(qbuf, kbuf, vT, geo_mask, sem_mask, attn);

  // ---- chebnet path ----
  k_asup<<<8, 256, 0, stream>>>(node_emb, Asup);
  k_za<<<128, 256, 0, stream>>>(Asup, zT, zA);
  k_cheb<<<64, 256, 16 * 128 * 40 * sizeof(__bf16), stream>>>(
      node_emb, WpT, zbf, zA, bias_pool, zstat);

  // ---- dynamic path GEMM chain ----
  k_gemm128<<<128, 256, 0, stream>>>(attn, Wbf + 49152, proj_b, nullptr,
                                     nullptr, proj_out, 0);
  k_gemm128<<<128, 256, 0, stream>>>(proj_out, Wbf + 65536, amlp_b1, nullptr,
                                     nullptr, amlp_h, 1);
  k_gemm128<<<128, 256, 0, stream>>>(amlp_h, Wbf + 81920, amlp_b2, zstat,
                                     z1, nullptr, 0);
  // ---- residual MLP ----
  k_ln<<<2048, 256, 0, stream>>>(z1, ln2_g, ln2_b, hn2, 16384);
  k_gemm128<<<128, 256, 0, stream>>>(hn2, Wbf + 98304, bmlp_b1, nullptr,
                                     nullptr, bmlp_h, 1);
  k_gemm128<<<128, 256, 0, stream>>>(bmlp_h, Wbf + 114688, bmlp_b2, z1,
                                     out, nullptr, 0);
}
